// CustomTransformerEncoderMoELayerStoich_61538291417558
// MI455X (gfx1250) — matrix operand order repaired and checked by reading, not compiled, not run on hardware
//
#include <hip/hip_runtime.h>
#include <hip/hip_bf16.h>

// ---------------- problem constants ----------------
#define Bb   4
#define Tt   1024
#define Dd   1024
#define Hh   16
#define HDd  64
#define FFf  4096
#define Ee   8
#define BT   (Bb * Tt)

typedef __attribute__((ext_vector_type(16))) __bf16 v16bf;
typedef __attribute__((ext_vector_type(8)))  float  v8f;
typedef __attribute__((ext_vector_type(4)))  unsigned int u32x4;
typedef __attribute__((ext_vector_type(8)))  int i32x8;
typedef __attribute__((ext_vector_type(4)))  int i32x4;

#define WMMA_BF16(a, b, c) \
  __builtin_amdgcn_wmma_f32_16x16x32_bf16(false, (a), false, (b), (short)0, (c), false, false)

__device__ __forceinline__ v8f v8f_zero() {
  v8f z = {0.f, 0.f, 0.f, 0.f, 0.f, 0.f, 0.f, 0.f};
  return z;
}

// ---- CDNA5 async global->LDS copy (ASYNCcnt-tracked DMA) ----
__device__ __forceinline__ void async_copy_b128(void* lds_dst, const void* gsrc) {
  uint32_t l = (uint32_t)(uintptr_t)lds_dst;   // flat addr low 32 bits = LDS offset
  uint64_t g = (uint64_t)(uintptr_t)gsrc;
  asm volatile("global_load_async_to_lds_b128 %0, %1, off"
               :
               : "v"(l), "v"(g)
               : "memory");
}
__device__ __forceinline__ void wait_async0() {
  asm volatile("s_wait_asynccnt 0x0" ::: "memory");
}

// ---- CDNA5 Tensor Data Mover: 2D f32 tile global->LDS (TENSORcnt-tracked) ----
#if __has_builtin(__builtin_amdgcn_tensor_load_to_lds)
#define HAVE_TDM 1
__device__ __forceinline__ void tdm_load_2d_f32(uint32_t lds_off, const float* gsrc,
                                                unsigned tensor_w, unsigned tensor_h,
                                                unsigned tile_w, unsigned tile_h,
                                                int row_stride) {
  uint64_t ga = (uint64_t)(uintptr_t)gsrc;
  // D# group 0: count=1 | lds_addr | global_addr[56:0] | type=2 ("image")
  u32x4 g0;
  g0[0] = 1u;
  g0[1] = lds_off;
  g0[2] = (uint32_t)(ga & 0xffffffffu);
  g0[3] = (uint32_t)((ga >> 32) & 0x01ffffffu) | (2u << 30);
  // D# group 1: data_size=4B, pad 4 DWORDs every 128 DWORDs (LDS stride 132 f32),
  // tensor_dim0/1, tile_dim0/1, tensor_dim0_stride.
  i32x8 g1;
  g1[0] = (int)((2u << 16) | (1u << 20) | (6u << 22) | (3u << 25));
  g1[1] = (int)((tensor_w & 0xffffu) << 16);
  g1[2] = (int)(((tensor_w >> 16) & 0xffffu) | ((tensor_h & 0xffffu) << 16));
  g1[3] = (int)(((tensor_h >> 16) & 0xffffu) | ((tile_w & 0xffffu) << 16));
  g1[4] = (int)(tile_h & 0xffffu);
  g1[5] = row_stride;
  g1[6] = 0;
  g1[7] = 0;
  i32x4 z4 = {0, 0, 0, 0};                 // groups 2/3 unused (2D tensor)
  i32x8 z8 = {0, 0, 0, 0, 0, 0, 0, 0};     // trailing group (toolchain 6-arg form)
  __builtin_amdgcn_tensor_load_to_lds(g0, g1, z4, z4, z8, 0);
}
#else
#define HAVE_TDM 0
#endif

__device__ __forceinline__ void wait_tensor0() {
#if __has_builtin(__builtin_amdgcn_s_wait_tensorcnt)
  __builtin_amdgcn_s_wait_tensorcnt(0);
#else
  asm volatile("s_wait_tensorcnt 0x0" ::: "memory");
#endif
}

// A-fragment (16x32 bf16, M x K). element (m,k) at t[m*stride + k].
__device__ __forceinline__ v16bf load_a_frag(const __bf16* t, int stride) {
  const int lane = threadIdx.x & 31;
  const int m = lane & 15;
  const int hlf = lane >> 4;
  v16bf a;
#pragma unroll
  for (int v = 0; v < 8; ++v) {
    int k = hlf * 16 + 2 * v;
    a[2 * v]     = t[m * stride + k];
    a[2 * v + 1] = t[m * stride + k + 1];
  }
  return a;
}

// B-fragment (32x16 bf16, K x N). element (k,n) at t[k*stride + n].
__device__ __forceinline__ v16bf load_b_frag(const __bf16* t, int stride) {
  const int lane = threadIdx.x & 31;
  const int n = lane & 15;
  const int khalf = (lane >> 4) << 4;
  v16bf b;
#pragma unroll
  for (int v = 0; v < 8; ++v) {
    int k = khalf + 2 * v;
    b[2 * v]     = t[k * stride + n];
    b[2 * v + 1] = t[(k + 1) * stride + n];
  }
  return b;
}

// B-fragment where element (k,n) sits at t[n*stride + k] (transposed source, e.g. K^T)
__device__ __forceinline__ v16bf load_bT_frag(const __bf16* t, int stride) {
  const int lane = threadIdx.x & 31;
  const int n = lane & 15;
  const int khalf = (lane >> 4) << 4;
  v16bf b;
#pragma unroll
  for (int v = 0; v < 8; ++v) {
    int k = khalf + 2 * v;
    b[2 * v]     = t[n * stride + k];
    b[2 * v + 1] = t[n * stride + k + 1];
  }
  return b;
}

// B-fragment gathered from f32 memory (global or LDS) with on-the-fly bf16 convert
__device__ __forceinline__ v16bf load_b_frag_f32(const float* t, int stride) {
  const int lane = threadIdx.x & 31;
  const int n = lane & 15;
  const int khalf = (lane >> 4) << 4;
  v16bf b;
#pragma unroll
  for (int v = 0; v < 8; ++v) {
    int k = khalf + 2 * v;
    b[2 * v]     = (__bf16)t[(size_t)k * stride + n];
    b[2 * v + 1] = (__bf16)t[(size_t)(k + 1) * stride + n];
  }
  return b;
}

// =====================================================================
// Generic 128x128-tile GEMM, bf16 WMMA with f32 accumulate.
// MODE 0: QKV projection (grid.z selects Wq/Wk/Wv), writes bf16 [B,H,T,HD]
// MODE 1: out-projection, adds bias + residual(src), writes f32 [BT,D]
// =====================================================================
template <int MODE>
__global__ __launch_bounds__(256) void gemm_kernel(
    const float* __restrict__ A, int K, int N,
    const float* __restrict__ Wa, const float* __restrict__ Wb,
    const float* __restrict__ Wc,
    const float* __restrict__ ba, const float* __restrict__ bbv,
    const float* __restrict__ bc,
    const float* __restrict__ resid, float* __restrict__ outF,
    __bf16* __restrict__ oq, __bf16* __restrict__ ok, __bf16* __restrict__ ov) {
  const int rowBase = blockIdx.x * 128;
  const int colBase = blockIdx.y * 128;
  const int z = blockIdx.z;
  const float* W    = (z == 0) ? Wa : (z == 1) ? Wb : Wc;
  const float* bias = (z == 0) ? ba : (z == 1) ? bbv : bc;
  __bf16* oqkv      = (z == 0) ? oq : (z == 1) ? ok : ov;

  const int tid = threadIdx.x;
  const int lane = tid & 31, wave = tid >> 5;
  const int hlf = lane >> 4, n = lane & 15;

  __shared__ __bf16 As[128 * 34];
  __shared__ __bf16 Bs[32 * 136];

  v8f acc[8];
#pragma unroll
  for (int j = 0; j < 8; ++j) acc[j] = v8f_zero();

#pragma unroll 1
  for (int k0 = 0; k0 < K; k0 += 32) {
    __syncthreads();
#pragma unroll
    for (int i = 0; i < 16; ++i) {
      int idx = tid + i * 256;
      int r = idx >> 5, c = idx & 31;
      As[r * 34 + c] = (__bf16)A[(size_t)(rowBase + r) * K + k0 + c];
    }
#pragma unroll
    for (int i = 0; i < 16; ++i) {
      int idx = tid + i * 256;
      int r = idx >> 7, c = idx & 127;
      Bs[r * 136 + c] = (__bf16)W[(size_t)(k0 + r) * N + colBase + c];
    }
    if (k0 + 32 < K) {  // prefetch next tiles into cache
      __builtin_prefetch(&A[(size_t)(rowBase + (tid >> 1)) * K + k0 + 32], 0, 1);
      __builtin_prefetch(&W[(size_t)(k0 + 32 + (tid & 31)) * N + colBase], 0, 1);
    }
    __syncthreads();
    v16bf a = load_a_frag(&As[wave * 16 * 34], 34);
#pragma unroll
    for (int j = 0; j < 8; ++j) {
      v16bf b = load_b_frag(&Bs[j * 16], 136);
      acc[j] = WMMA_BF16(a, b, acc[j]);
    }
  }

#pragma unroll
  for (int j = 0; j < 8; ++j) {
#pragma unroll
    for (int r = 0; r < 8; ++r) {
      int row = rowBase + wave * 16 + r + hlf * 8;
      int col = colBase + j * 16 + n;
      float val = acc[j][r] + bias[col];
      if (MODE == 0) {
        int bI = row >> 10, t = row & (Tt - 1);
        int hI = col >> 6, hd = col & (HDd - 1);
        oqkv[(((size_t)bI * Hh + hI) * Tt + t) * HDd + hd] = (__bf16)val;
      } else {
        val += resid[(size_t)row * N + col];
        outF[(size_t)row * N + col] = val;
      }
    }
  }
}

// =====================================================================
// Flash attention with per-head stoichiometric bias.
// grid: (B*H, T/128). Each wave owns 16 query rows; streams 32-key tiles
// through double-buffered LDS using async global->LDS DMA (ASYNCcnt).
// =====================================================================
__global__ __launch_bounds__(256) void flash_kernel(
    const __bf16* __restrict__ qb, const __bf16* __restrict__ kb,
    const __bf16* __restrict__ vb, const float* __restrict__ frac,
    const float* __restrict__ apos, const float* __restrict__ aneg,
    float* __restrict__ attnOut) {
  const int bh = blockIdx.x;
  const int bI = bh >> 4, hI = bh & 15;
  const int qbase = blockIdx.y * 128;
  const int tid = threadIdx.x, lane = tid & 31, wave = tid >> 5;
  const int hlf = lane >> 4, n = lane & 15;

  // stride 72 elems = 144 B (16B multiple) so every 16B async chunk lands aligned
  __shared__ alignas(16) __bf16 kt[2][32][72];
  __shared__ alignas(16) __bf16 vt[2][32][72];
  __shared__ alignas(16) __bf16 pt[8][16][36];
  __shared__ float fq[128];
  __shared__ float fk[2][32];

  const size_t bhT = ((size_t)bI * Hh + hI) * Tt;
  const __bf16* qp = qb + (bhT + qbase + (size_t)wave * 16) * HDd;

  v16bf qa0 = load_a_frag(qp, HDd);       // d 0..31
  v16bf qa1 = load_a_frag(qp + 32, HDd);  // d 32..63

  if (tid < 128) fq[tid] = frac[(size_t)bI * Tt + qbase + tid];

  const float scale = 0.125f;  // HD^-0.5
  const float ap = apos[hI], an = aneg[hI];

  // each thread DMAs one 16B chunk of K and one of V per tile
  const int sr = tid >> 3, sc = (tid & 7) * 8;
  auto issue_tile = [&](int buf, int kt0) {
    async_copy_b128(&kt[buf][sr][sc], kb + (bhT + kt0 + sr) * HDd + sc);
    async_copy_b128(&vt[buf][sr][sc], vb + (bhT + kt0 + sr) * HDd + sc);
    if (tid < 32) fk[buf][tid] = frac[(size_t)bI * Tt + kt0 + tid];
  };

  float mrow[8], lrow[8];
  v8f o[4];
#pragma unroll
  for (int r = 0; r < 8; ++r) { mrow[r] = -3.0e38f; lrow[r] = 0.f; }
#pragma unroll
  for (int t4 = 0; t4 < 4; ++t4) o[t4] = v8f_zero();

  issue_tile(0, 0);  // prologue DMA

#pragma unroll 1
  for (int kt0 = 0; kt0 < Tt; kt0 += 32) {
    const int buf = (kt0 >> 5) & 1;
    wait_async0();   // this wave's DMA into `buf` landed
    __syncthreads(); // everyone's DMA into `buf` landed
    if (kt0 + 32 < Tt) issue_tile(buf ^ 1, kt0 + 32);  // overlap next tile

    // S = Q K^T : two 16x16 key tiles, K-dim = 64
    v8f s0 = v8f_zero(), s1 = v8f_zero();
    s0 = WMMA_BF16(qa0, load_bT_frag(&kt[buf][0][0], 72), s0);
    s0 = WMMA_BF16(qa1, load_bT_frag(&kt[buf][0][32], 72), s0);
    s1 = WMMA_BF16(qa0, load_bT_frag(&kt[buf][16][0], 72), s1);
    s1 = WMMA_BF16(qa1, load_bT_frag(&kt[buf][16][32], 72), s1);

    float alr[8];
#pragma unroll
    for (int r = 0; r < 8; ++r) {
      int row = r + hlf * 8;
      float fi = fq[wave * 16 + row];
      float d0 = fk[buf][n] - fi, d1 = fk[buf][n + 16] - fi;
      float v0 = s0[r] * scale + (d0 > 0.f ? ap * d0 : an * d0);
      float v1 = s1[r] * scale + (d1 > 0.f ? ap * d1 : an * d1);
      float mx = fmaxf(v0, v1);
      mx = fmaxf(mx, __shfl_xor(mx, 1));
      mx = fmaxf(mx, __shfl_xor(mx, 2));
      mx = fmaxf(mx, __shfl_xor(mx, 4));
      mx = fmaxf(mx, __shfl_xor(mx, 8));
      float mn = fmaxf(mrow[r], mx);
      float al = __expf(mrow[r] - mn);
      float p0 = __expf(v0 - mn), p1 = __expf(v1 - mn);
      float sm = p0 + p1;
      sm += __shfl_xor(sm, 1);
      sm += __shfl_xor(sm, 2);
      sm += __shfl_xor(sm, 4);
      sm += __shfl_xor(sm, 8);
      lrow[r] = lrow[r] * al + sm;
      mrow[r] = mn;
      alr[r] = al;
      pt[wave][row][n]      = (__bf16)p0;
      pt[wave][row][n + 16] = (__bf16)p1;
    }
#pragma unroll
    for (int t4 = 0; t4 < 4; ++t4)
#pragma unroll
      for (int r = 0; r < 8; ++r) o[t4][r] *= alr[r];

    // O += P V  (K-dim = 32 keys, 4 HD-column tiles)
    v16bf pa = load_a_frag(&pt[wave][0][0], 36);
#pragma unroll
    for (int t4 = 0; t4 < 4; ++t4) {
      v16bf bvf = load_b_frag(&vt[buf][0][t4 * 16], 72);
      o[t4] = WMMA_BF16(pa, bvf, o[t4]);
    }
    __syncthreads();  // all reads of `buf` done before it is DMA-overwritten
  }

#pragma unroll
  for (int t4 = 0; t4 < 4; ++t4)
#pragma unroll
    for (int r = 0; r < 8; ++r) {
      int row = qbase + wave * 16 + r + hlf * 8;
      int col = hI * HDd + t4 * 16 + n;
      attnOut[((size_t)bI * Tt + row) * Dd + col] = o[t4][r] / lrow[r];
    }
}

// =====================================================================
// LayerNorm over D=1024, one block per row; optionally duplicates output
// (used to seed the residual accumulator for the MoE atomics).
// =====================================================================
__global__ __launch_bounds__(256) void ln_kernel(
    const float* __restrict__ in, const float* __restrict__ g,
    const float* __restrict__ bta, float* __restrict__ out,
    float* __restrict__ dup) {
  const int row = blockIdx.x;
  const int tid = threadIdx.x;
  const float* x = in + (size_t)row * Dd;
  float v[4], s = 0.f, ss = 0.f;
#pragma unroll
  for (int i = 0; i < 4; ++i) {
    v[i] = x[tid + i * 256];
    s += v[i];
    ss += v[i] * v[i];
  }
#pragma unroll
  for (int m = 1; m < 32; m <<= 1) {
    s += __shfl_xor(s, m);
    ss += __shfl_xor(ss, m);
  }
  __shared__ float rs[8], rss[8];
  if ((tid & 31) == 0) { rs[tid >> 5] = s; rss[tid >> 5] = ss; }
  __syncthreads();
  s = rs[tid & 7];
  ss = rss[tid & 7];
#pragma unroll
  for (int m = 1; m < 8; m <<= 1) {
    s += __shfl_xor(s, m);
    ss += __shfl_xor(ss, m);
  }
  float mu = s * (1.f / Dd);
  float var = ss * (1.f / Dd) - mu * mu;
  float inv = rsqrtf(var + 1e-5f);
#pragma unroll
  for (int i = 0; i < 4; ++i) {
    int c = tid + i * 256;
    float y = (v[i] - mu) * inv * g[c] + bta[c];
    out[(size_t)row * Dd + c] = y;
    if (dup) dup[(size_t)row * Dd + c] = y;
  }
}

// =====================================================================
// Gate: logits -> softmax -> top2 -> dense weights + per-expert token lists.
// =====================================================================
__global__ __launch_bounds__(256) void gate_kernel(
    const float* __restrict__ x1, const float* __restrict__ gW,
    const float* __restrict__ gb, float* __restrict__ gatew,
    int* __restrict__ counts, int* __restrict__ lists) {
  const int tid = threadIdx.x, lane = tid & 31, wave = tid >> 5;
  const int tok = blockIdx.x * 8 + wave;
  const int ee = lane & 7, gq = lane >> 3;
  const float* xr = x1 + (size_t)tok * Dd;
  float acc = 0.f;
#pragma unroll 1
  for (int d = gq * 256; d < gq * 256 + 256; ++d) acc += xr[d] * gW[d * Ee + ee];
  acc += __shfl_xor(acc, 8);
  acc += __shfl_xor(acc, 16);
  float logit = acc + gb[ee];
  float mx = logit;
  mx = fmaxf(mx, __shfl_xor(mx, 1));
  mx = fmaxf(mx, __shfl_xor(mx, 2));
  mx = fmaxf(mx, __shfl_xor(mx, 4));
  float ex = __expf(logit - mx);
  float sm = ex;
  sm += __shfl_xor(sm, 1);
  sm += __shfl_xor(sm, 2);
  sm += __shfl_xor(sm, 4);
  float sc = ex / sm;
  float bs = sc;
  int bi = ee;
#pragma unroll
  for (int m = 1; m < 8; m <<= 1) {
    float os = __shfl_xor(bs, m);
    int oi = __shfl_xor(bi, m);
    if (os > bs || (os == bs && oi < bi)) { bs = os; bi = oi; }
  }
  float s2 = (ee == bi) ? -1.f : sc;
  float bs2 = s2;
  int bi2 = ee;
#pragma unroll
  for (int m = 1; m < 8; m <<= 1) {
    float os = __shfl_xor(bs2, m);
    int oi = __shfl_xor(bi2, m);
    if (os > bs2 || (os == bs2 && oi < bi2)) { bs2 = os; bi2 = oi; }
  }
  if (gq == 0)
    gatew[(size_t)tok * Ee + ee] = (ee == bi) ? bs : (ee == bi2) ? bs2 : 0.f;
  if (lane == 0) {
    int p1 = atomicAdd(&counts[bi], 1);
    lists[bi * BT + p1] = tok;
    int p2 = atomicAdd(&counts[bi2], 1);
    lists[bi2 * BT + p2] = tok;
  }
}

// =====================================================================
// Routed MoE expert kernel. grid = (BT/16, E). W1 tiles arrive via the
// Tensor Data Mover (TENSORcnt DMA) when available; W2 fragments are read
// directly from global (no inter-wave reuse). Results scatter with f32 atomics.
// =====================================================================
__global__ __launch_bounds__(256) void moe_kernel(
    const float* __restrict__ x1, const float* __restrict__ W1g,
    const float* __restrict__ b1g, const float* __restrict__ W2g,
    const float* __restrict__ b2g, const float* __restrict__ gatew,
    const int* __restrict__ counts, const int* __restrict__ lists,
    float* __restrict__ x2raw) {
  const int e = blockIdx.y;
  const int cnt = counts[e];
  const int tbase = blockIdx.x * 16;
  if (tbase >= cnt) return;
  const int tid = threadIdx.x, lane = tid & 31, wave = tid >> 5;
  const int hlf = lane >> 4, n = lane & 15;

  __shared__ int toks[16];
  __shared__ float wts[16];
  __shared__ __bf16 xt[16][1040];
  __shared__ alignas(16) float w1f[32][132];  // TDM-padded f32 stage (stride 132)
  __shared__ __bf16 ht[16][136];

  if (tid < 16) {
    int i = tbase + tid;
    int tok = (i < cnt) ? lists[e * BT + i] : 0;
    toks[tid] = tok;
    wts[tid] = (i < cnt) ? gatew[(size_t)tok * Ee + e] : 0.f;
  }
  __syncthreads();
#pragma unroll 1
  for (int i = 0; i < 64; ++i) {
    int idx = tid + i * 256;
    int r = idx >> 10, c = idx & 1023;
    xt[r][c] = (__bf16)x1[(size_t)toks[r] * Dd + c];
  }
  __syncthreads();

  const float* W1e = W1g + (size_t)e * Dd * FFf;
  const float* W2e = W2g + (size_t)e * FFf * Dd;
  const float* b1e = b1g + e * FFf;
  const float* b2e = b2g + e * Dd;

  v8f o[8];
#pragma unroll
  for (int j = 0; j < 8; ++j) o[j] = v8f_zero();

#pragma unroll 1
  for (int f0 = 0; f0 < FFf; f0 += 128) {
    // phase 1: h_chunk[16,128] = relu(x @ W1e[:, f0:f0+128] + b1)
    v8f hacc = v8f_zero();
#pragma unroll 1
    for (int k0 = 0; k0 < Dd; k0 += 32) {
      __syncthreads();
#if HAVE_TDM
      if (wave == 0) {
        tdm_load_2d_f32((uint32_t)(uintptr_t)&w1f[0][0],
                        W1e + (size_t)k0 * FFf + f0,
                        FFf, Dd, 128, 32, FFf);
        wait_tensor0();
      }
#else
#pragma unroll
      for (int i = 0; i < 16; ++i) {
        int idx = tid + i * 256;
        int r = idx >> 7, c = idx & 127;
        w1f[r][c] = W1e[(size_t)(k0 + r) * FFf + f0 + c];
      }
      if (k0 + 32 < Dd)
        __builtin_prefetch(&W1e[(size_t)(k0 + 32 + (tid & 31)) * FFf + f0], 0, 1);
#endif
      __syncthreads();
      v16bf a = load_a_frag(&xt[0][k0], 1040);
      v16bf b = load_b_frag_f32(&w1f[0][wave * 16], 132);
      hacc = WMMA_BF16(a, b, hacc);
    }
    __syncthreads();
#pragma unroll
    for (int r = 0; r < 8; ++r) {
      int row = r + hlf * 8;
      float hv = hacc[r] + b1e[f0 + wave * 16 + n];
      ht[row][wave * 16 + n] = (__bf16)fmaxf(hv, 0.f);
    }
    __syncthreads();
    // phase 2: o += h_chunk @ W2e[f0:f0+128, wave's 128 cols]
#pragma unroll
    for (int k2 = 0; k2 < 128; k2 += 32) {
      v16bf a = load_a_frag(&ht[0][k2], 136);
#pragma unroll
      for (int j = 0; j < 8; ++j) {
        v16bf b =
            load_b_frag_f32(W2e + (size_t)(f0 + k2) * Dd + wave * 128 + j * 16, Dd);
        o[j] = WMMA_BF16(a, b, o[j]);
      }
    }
  }

#pragma unroll
  for (int j = 0; j < 8; ++j)
#pragma unroll
    for (int r = 0; r < 8; ++r) {
      int row = r + hlf * 8;
      int col = wave * 128 + j * 16 + n;
      float val = (o[j][r] + b2e[col]) * wts[row];
      atomicAdd(&x2raw[(size_t)toks[row] * Dd + col], val);
    }
}

// =====================================================================
extern "C" void kernel_launch(void* const* d_in, const int* in_sizes, int n_in,
                              void* d_out, int out_size, void* d_ws,
                              size_t ws_size, hipStream_t stream) {
  (void)in_sizes; (void)n_in; (void)out_size; (void)ws_size;
  const float* src  = (const float*)d_in[0];
  const float* frac = (const float*)d_in[1];
  const float* Wq = (const float*)d_in[2];
  const float* bq = (const float*)d_in[3];
  const float* Wk = (const float*)d_in[4];
  const float* bk = (const float*)d_in[5];
  const float* Wv = (const float*)d_in[6];
  const float* bv = (const float*)d_in[7];
  const float* Wo = (const float*)d_in[8];
  const float* bo = (const float*)d_in[9];
  const float* apos = (const float*)d_in[10];
  const float* aneg = (const float*)d_in[11];
  const float* gW = (const float*)d_in[12];
  const float* gb = (const float*)d_in[13];
  const float* W1 = (const float*)d_in[14];
  const float* b1 = (const float*)d_in[15];
  const float* W2 = (const float*)d_in[16];
  const float* b2 = (const float*)d_in[17];
  const float* ln1g = (const float*)d_in[18];
  const float* ln1b = (const float*)d_in[19];
  const float* ln2g = (const float*)d_in[20];
  const float* ln2b = (const float*)d_in[21];

  char* ws = (char*)d_ws;
  size_t off = 0;
  auto alloc = [&](size_t bytes) -> void* {
    void* p = ws + off;
    off += (bytes + 255) & ~(size_t)255;
    return p;
  };
  __bf16* qb = (__bf16*)alloc(sizeof(__bf16) * (size_t)BT * Dd);
  __bf16* kb = (__bf16*)alloc(sizeof(__bf16) * (size_t)BT * Dd);
  __bf16* vb = (__bf16*)alloc(sizeof(__bf16) * (size_t)BT * Dd);
  float* attn  = (float*)alloc(sizeof(float) * (size_t)BT * Dd);
  float* pre1  = (float*)alloc(sizeof(float) * (size_t)BT * Dd);
  float* x1    = (float*)alloc(sizeof(float) * (size_t)BT * Dd);
  float* x2raw = (float*)alloc(sizeof(float) * (size_t)BT * Dd);
  float* gatew = (float*)alloc(sizeof(float) * (size_t)BT * Ee);
  int* counts  = (int*)alloc(sizeof(int) * Ee);
  int* lists   = (int*)alloc(sizeof(int) * (size_t)Ee * BT);

  (void)hipMemsetAsync(counts, 0, sizeof(int) * Ee, stream);

  // 1) QKV projections -> bf16 [B,H,T,HD]
  gemm_kernel<0><<<dim3(BT / 128, Dd / 128, 3), 256, 0, stream>>>(
      src, Dd, Dd, Wq, Wk, Wv, bq, bk, bv, nullptr, nullptr, qb, kb, vb);

  // 2) flash attention with stoichiometric bias -> attn f32 [BT,D]
  flash_kernel<<<dim3(Bb * Hh, Tt / 128), 256, 0, stream>>>(
      qb, kb, vb, frac, apos, aneg, attn);

  // 3) output projection + residual -> pre1
  gemm_kernel<1><<<dim3(BT / 128, Dd / 128, 1), 256, 0, stream>>>(
      attn, Dd, Dd, Wo, nullptr, nullptr, bo, nullptr, nullptr, src, pre1,
      nullptr, nullptr, nullptr);

  // 4) LN1 -> x1, and seed x2raw = x1 for MoE atomic accumulation
  ln_kernel<<<BT, 256, 0, stream>>>(pre1, ln1g, ln1b, x1, x2raw);

  // 5) gate: top-2 routing
  gate_kernel<<<BT / 8, 256, 0, stream>>>(x1, gW, gb, gatew, counts, lists);

  // 6) routed MoE experts: x2raw += w_e * FFN_e(x1)
  moe_kernel<<<dim3(BT / 16, Ee), 256, 0, stream>>>(
      x1, W1, b1, W2, b2, gatew, counts, lists, x2raw);

  // 7) LN2 -> final output
  ln_kernel<<<BT, 256, 0, stream>>>(x2raw, ln2g, ln2b, (float*)d_out, nullptr);
}
